// AriaGroupedExpertsGemm_6871947674155
// MI455X (gfx1250) — compile-verified, matrix operand order checked by reading, not run on hardware
//
#include <hip/hip_runtime.h>
#include <hip/hip_bf16.h>

// ---------------------------------------------------------------------------
// Grouped GEMM (MoE experts over contiguous token ranges) for gfx1250.
// f32 in / f32 out; math via V_WMMA_F32_16X16X32_BF16 (bf16 A/B, f32 accum).
// Software-pipelined: next K-chunk's global loads overlap current chunk's
// WMMAs; double-buffered LDS; packed f32->bf16 converts via convertvector
// (lowers to v_cvt_pk_bf16_f32 when the target has it; RNE either way).
// ---------------------------------------------------------------------------

typedef __attribute__((ext_vector_type(16))) __bf16 v16bf;
typedef __attribute__((ext_vector_type(8)))  __bf16 v8bf;
typedef __attribute__((ext_vector_type(2)))  __bf16 v2bf;
typedef __attribute__((ext_vector_type(2)))  float  v2f;
typedef __attribute__((ext_vector_type(8)))  float  v8f;

#define BM 128
#define BN 128
#define BK 64          // K-chunk (f32 in global, bf16 in LDS)
#define BKP 72         // padded LDS stride (bf16 elems): 144B
#define NTHREADS 256   // 8 waves (wave32)

union FragBF16 { v16bf v; v8bf h[2]; };

// two f32 -> packed 2x bf16 in one dword, round-to-nearest-even (fptrunc);
// backend selects the packed hardware convert where available
__device__ __forceinline__ unsigned pack2_bf16(float lo, float hi) {
    v2f f = {lo, hi};
    v2bf r = __builtin_convertvector(f, v2bf);
    return __builtin_bit_cast(unsigned, r);
}

__global__ __launch_bounds__(NTHREADS, 2)
void aria_grouped_gemm_wmma(const float* __restrict__ x,
                            const float* __restrict__ w,
                            const int*   __restrict__ tokens_per_expert,
                            float*       __restrict__ out,
                            int K, int N, int n_experts)
{
    __shared__ unsigned short ldsA[2][BM * BKP];   // row-major, stride BKP
    __shared__ unsigned short ldsB[2][BN * BKP];   // TRANSPOSED: n-major

    const int t    = threadIdx.x;
    const int lane = t & 31;
    const int wid  = t >> 5;
    const int half = lane >> 4;
    const int l16  = lane & 15;

    const int m0 = blockIdx.y * BM;
    const int n0 = blockIdx.x * BN;

    // expert owning this row tile (group sizes are multiples of 128, so a
    // 128-row tile never straddles an expert boundary)
    int e = 0;
    {
        int start = 0;
        for (int i = 0; i < n_experts; ++i) {
            int c = tokens_per_expert[i];
            if (m0 < start + c) { e = i; break; }
            start += c;
        }
    }
    const float* __restrict__ wE = w + (size_t)e * (size_t)K * (size_t)N;

    // wave tiling: 4 (M) x 2 (N); each wave owns 32x64 = 2x4 WMMA tiles
    const int waveM = (wid & 3) * 32;
    const int waveN = (wid >> 2) * 64;

    v8f acc[2][4] = {};

    // ---- register staging for the software pipeline ----
    float4 ra[8];            // A: 128x64 f32 -> 2048 float4 / 256 thr = 8
    float4 rb0[4], rb1[4];   // B: pairs of K-adjacent rows, same 4 columns

    auto issue_loads = [&](int k0) {
        #pragma unroll
        for (int i = 0; i < 8; ++i) {
            int id  = t + i * NTHREADS;     // 0..2047
            int row = id >> 4;              // 16 float4 per A row
            int c4  = id & 15;
            ra[i] = *(const float4*)(x + (size_t)(m0 + row) * K + k0 + c4 * 4);
        }
        #pragma unroll
        for (int i = 0; i < 4; ++i) {
            int id = t + i * NTHREADS;      // 0..1023
            int kp = id >> 5;               // 0..31 -> k = 2*kp
            int n4 = id & 31;
            int k  = k0 + 2 * kp;
            rb0[i] = *(const float4*)(wE + (size_t)k       * N + n0 + n4 * 4);
            rb1[i] = *(const float4*)(wE + (size_t)(k + 1) * N + n0 + n4 * 4);
        }
    };

    auto store_lds = [&](int buf) {
        #pragma unroll
        for (int i = 0; i < 8; ++i) {
            int id  = t + i * NTHREADS;
            int row = id >> 4;
            int c4  = id & 15;
            uint2 p;
            p.x = pack2_bf16(ra[i].x, ra[i].y);
            p.y = pack2_bf16(ra[i].z, ra[i].w);
            *(uint2*)&ldsA[buf][row * BKP + c4 * 4] = p;
        }
        #pragma unroll
        for (int i = 0; i < 4; ++i) {
            int id = t + i * NTHREADS;
            int kp = id >> 5;
            int n4 = id & 31;
            int k  = 2 * kp;               // even -> dword-aligned LDS store
            *(unsigned*)&ldsB[buf][(n4 * 4 + 0) * BKP + k] = pack2_bf16(rb0[i].x, rb1[i].x);
            *(unsigned*)&ldsB[buf][(n4 * 4 + 1) * BKP + k] = pack2_bf16(rb0[i].y, rb1[i].y);
            *(unsigned*)&ldsB[buf][(n4 * 4 + 2) * BKP + k] = pack2_bf16(rb0[i].z, rb1[i].z);
            *(unsigned*)&ldsB[buf][(n4 * 4 + 3) * BKP + k] = pack2_bf16(rb0[i].w, rb1[i].w);
        }
    };

    const int nChunks = K / BK;
    issue_loads(0);

    for (int kc = 0; kc < nChunks; ++kc) {
        const int buf = kc & 1;

        // drain chunk kc from registers into LDS buffer `buf`
        store_lds(buf);

        // start chunk kc+1's global loads (overlap with WMMAs below)
        if (kc + 1 < nChunks) {
            issue_loads((kc + 1) * BK);
            if (kc + 2 < nChunks) {
                // pull chunk kc+2's lines toward L2 (global_prefetch_b8)
                int k2 = (kc + 2) * BK;
                __builtin_prefetch(x + (size_t)(m0 + (t >> 1)) * K + k2 + (t & 1) * 32, 0, 0);
                __builtin_prefetch(wE + (size_t)(k2 + (t >> 2)) * N + n0 + (t & 3) * 32, 0, 0);
            }
        }

        __syncthreads();   // stores to `buf` visible; also fences reuse of `buf`

        // ---- compute on buffer `buf`: 2 k-steps of 32, 8 WMMAs each ----
        #pragma unroll
        for (int kk = 0; kk < BK; kk += 32) {
            FragBF16 aF[2], bF[4];
            // 16-bit A/B layout: low lanes K 0-7 / 16-23, high lanes 8-15 / 24-31
            #pragma unroll
            for (int im = 0; im < 2; ++im) {
                const unsigned short* p =
                    &ldsA[buf][(waveM + im * 16 + l16) * BKP + kk + half * 8];
                aF[im].h[0] = *(const v8bf*)p;
                aF[im].h[1] = *(const v8bf*)(p + 16);
            }
            #pragma unroll
            for (int in = 0; in < 4; ++in) {
                const unsigned short* p =
                    &ldsB[buf][(waveN + in * 16 + l16) * BKP + kk + half * 8];
                bF[in].h[0] = *(const v8bf*)p;
                bF[in].h[1] = *(const v8bf*)(p + 16);
            }
            #pragma unroll
            for (int im = 0; im < 2; ++im)
                #pragma unroll
                for (int in = 0; in < 4; ++in)
                    acc[im][in] = __builtin_amdgcn_wmma_f32_16x16x32_bf16(
                        false, aF[im].v, false, bF[in].v,
                        (short)0, acc[im][in], false, false);
        }
    }

    // ---- epilogue: 16x16 f32 C/D layout (VGPR r: M=r | M=8+r; N=lane%16) ----
    #pragma unroll
    for (int im = 0; im < 2; ++im) {
        #pragma unroll
        for (int in = 0; in < 4; ++in) {
            const int col = n0 + waveN + in * 16 + l16;
            #pragma unroll
            for (int r = 0; r < 8; ++r) {
                const int row = m0 + waveM + im * 16 + half * 8 + r;
                out[(size_t)row * N + col] = acc[im][in][r];
            }
        }
    }
}

extern "C" void kernel_launch(void* const* d_in, const int* in_sizes, int n_in,
                              void* d_out, int out_size, void* d_ws, size_t ws_size,
                              hipStream_t stream) {
    const float* x   = (const float*)d_in[0];
    const float* w   = (const float*)d_in[1];
    const int*   tpe = (const int*)d_in[2];
    float*       out = (float*)d_out;

    const int K = 2048;
    const int N = 4096;
    const int E = 8;
    const int M = in_sizes[0] / K;   // 8192

    dim3 grid(N / BN, M / BM);       // 32 x 64 blocks
    dim3 block(NTHREADS);
    aria_grouped_gemm_wmma<<<grid, block, 0, stream>>>(x, w, tpe, out, K, N, E);
}
